// AssemblyQueryEncoder_49538152792352
// MI455X (gfx1250) — matrix operand order repaired and checked by reading, not compiled, not run on hardware
//
#include <hip/hip_runtime.h>
#include <hip/hip_bf16.h>
#include <math.h>

typedef __bf16 bf16_t;
typedef __attribute__((ext_vector_type(16))) __bf16 v16bf;
typedef __attribute__((ext_vector_type(8)))  __bf16 v8bf;
typedef __attribute__((ext_vector_type(8)))  float  v8f;

#define N_NODES 40000
#define N_EDGES 640000
#define HID     128
#define OUTD    64
#define NGRAPH  64
#define SA_STRIDE 144   // padded LDS row stride (elements) to spread b128 reads over banks

// ---------------- init / utility kernels ----------------

__global__ void gcn_init_kernel(float* __restrict__ deg, float* __restrict__ pooled,
                                float* __restrict__ cnt) {
    int i = blockIdx.x * blockDim.x + threadIdx.x;
    if (i < N_NODES) deg[i] = 1.0f;              // self-loop contribution
    if (i < NGRAPH * HID) pooled[i] = 0.0f;
    if (i < NGRAPH) cnt[i] = 0.0f;
}

__global__ void gcn_zero_f32(float* __restrict__ p, int n) {
    int i = blockIdx.x * blockDim.x + threadIdx.x;
    if (i < n) p[i] = 0.0f;
}

__global__ void gcn_edge_deg(const long long* __restrict__ dst, float* __restrict__ deg) {
    int e = blockIdx.x * blockDim.x + threadIdx.x;
    if (e < N_EDGES) atomicAdd(&deg[(int)dst[e]], 1.0f);
}

__global__ void gcn_dinv(const float* __restrict__ deg, float* __restrict__ dinv) {
    int i = blockIdx.x * blockDim.x + threadIdx.x;
    if (i < N_NODES) dinv[i] = rsqrtf(deg[i]);
}

__global__ void gcn_cvt_bf16(const float* __restrict__ in, bf16_t* __restrict__ out, int n) {
    int i = blockIdx.x * blockDim.x + threadIdx.x;
    if (i < n) out[i] = (bf16_t)in[i];
}

// ---------------- pack W [128,128] f32 -> bf16 B-fragment order ----------------
// Wp[(((ct*4 + kb)*32 + lane)*16 + j] = W[kb*32 + (lane<16?0:8) + (j<8?j:j+8)][ct*16 + (lane&15)]
__global__ void gcn_pack_w(const float* __restrict__ W, bf16_t* __restrict__ Wp) {
    int idx = blockIdx.x * blockDim.x + threadIdx.x;
    if (idx >= 8 * 4 * 32 * 16) return;
    int j  = idx & 15;
    int l  = (idx >> 4) & 31;
    int kb = (idx >> 9) & 3;
    int ct = idx >> 11;
    int k  = kb * 32 + ((l < 16) ? 0 : 8) + ((j < 8) ? j : j + 8);
    int c  = ct * 16 + (l & 15);
    Wp[idx] = (bf16_t)W[(size_t)k * HID + c];
}

// ---------------- WMMA GEMM: out[N,128] = A[N,128] @ W[128,128] (fp32 acc) ----------------
// grid.x = row tiles (N/16); block = 256 threads = 8 waves; wave w owns column tile w.
// A tile staged in LDS once per block; B pre-packed so each lane loads 32B contiguous.
__global__ void gcn_gemm_wmma(const bf16_t* __restrict__ A, const bf16_t* __restrict__ Wp,
                              float* __restrict__ out) {
    __shared__ bf16_t sA[16 * SA_STRIDE];
    const int t    = threadIdx.x;
    const int wave = t >> 5;            // 0..7 -> column tile
    const int lane = t & 31;
    const int mtile = blockIdx.x;

    // cooperative A-tile load: 16 rows x 128 cols bf16 (4 KB), 16 B per thread
    {
        int row = t >> 4;               // 0..15
        int cc  = (t & 15) * 8;         // 0..120
        v8bf v = *(const v8bf*)(A + (size_t)(mtile * 16 + row) * HID + cc);
        *(v8bf*)(sA + row * SA_STRIDE + cc) = v;
    }
    __syncthreads();

    const int m      = lane & 15;       // row within tile (both lane halves)
    const int col    = wave * 16 + m;   // B/D column
    const int khalfA = (lane < 16) ? 0 : 8;   // A: K {0-7,16-23} vs {8-15,24-31}

    v8f acc = {};
#pragma unroll
    for (int kbi = 0; kbi < 4; ++kbi) {
        const bf16_t* ap = sA + m * SA_STRIDE + kbi * 32 + khalfA;
        v8bf lo = *(const v8bf*)ap;
        v8bf hi = *(const v8bf*)(ap + 16);
        v16bf af = __builtin_shufflevector(lo, hi,
                       0, 1, 2, 3, 4, 5, 6, 7, 8, 9, 10, 11, 12, 13, 14, 15);
        v16bf bfm = *(const v16bf*)(Wp + (((size_t)wave * 4 + kbi) * 32 + lane) * 16);
        acc = __builtin_amdgcn_wmma_f32_16x16x32_bf16(
            /*neg_a=*/false, af, /*neg_b=*/false, bfm,
            /*c_mod=*/(short)0, acc, /*reuse_a=*/false, /*reuse_b=*/false);
    }
    const int rbase = mtile * 16 + ((lane < 16) ? 0 : 8);  // D: VGPR r -> row r / r+8
#pragma unroll
    for (int r = 0; r < 8; ++r)
        out[(size_t)(rbase + r) * HID + col] = acc[r];
}

// ---------------- edge scatter: agg[dst] += h[src] * dinv[src]*dinv[dst] ----------------
// 32 threads per edge, float4 gather per thread
__global__ void gcn_scatter(const long long* __restrict__ src, const long long* __restrict__ dst,
                            const float* __restrict__ dinv, const float* __restrict__ h,
                            float* __restrict__ agg) {
    long long tid = (long long)blockIdx.x * blockDim.x + threadIdx.x;
    int e = (int)(tid >> 5);
    int f = ((int)tid & 31) * 4;
    if (e >= N_EDGES) return;
    int s = (int)src[e], d = (int)dst[e];
    float w = dinv[s] * dinv[d];
    float4 hv = *(const float4*)(h + (size_t)s * HID + f);
    float* ap = agg + (size_t)d * HID + f;
    atomicAdd(ap + 0, hv.x * w);
    atomicAdd(ap + 1, hv.y * w);
    atomicAdd(ap + 2, hv.z * w);
    atomicAdd(ap + 3, hv.w * w);
}

// ---------------- combine: out = relu(agg + dinv^2 * h + bias); also emit bf16 ----------------
__global__ void gcn_combine(const float* __restrict__ agg, const float* __restrict__ h,
                            const float* __restrict__ dinv, const float* __restrict__ bias,
                            float* __restrict__ out, bf16_t* __restrict__ outb) {
    long long tid = (long long)blockIdx.x * blockDim.x + threadIdx.x;
    if (tid >= (long long)N_NODES * HID) return;
    int i = (int)(tid >> 7);
    int f = (int)(tid & 127);
    float di = dinv[i];
    float v = agg[tid] + di * di * h[tid] + bias[f];
    float r = fmaxf(v, 0.0f);
    out[tid]  = r;
    outb[tid] = (bf16_t)r;
}

// ---------------- mean-pool accumulation ----------------
__global__ void gcn_pool(const float* __restrict__ feat, const long long* __restrict__ batch,
                         float* __restrict__ pooled, float* __restrict__ cnt) {
    long long tid = (long long)blockIdx.x * blockDim.x + threadIdx.x;
    if (tid >= (long long)N_NODES * HID) return;
    int i = (int)(tid >> 7);
    int f = (int)(tid & 127);
    int g = (int)batch[i];
    atomicAdd(&pooled[(size_t)g * HID + f], feat[tid]);
    if (f == 0) atomicAdd(&cnt[g], 1.0f);
}

// ---------------- head: mean -> @Wl + bl -> L2 normalize rows ----------------
// grid = 64 graphs, block = 64 threads (one per output column)
__global__ void gcn_head(const float* __restrict__ pooled, const float* __restrict__ cnt,
                         const float* __restrict__ Wl, const float* __restrict__ bl,
                         float* __restrict__ out) {
    __shared__ float mean[HID];
    __shared__ float red[OUTD];
    int g = blockIdx.x;
    int j = threadIdx.x;  // 0..63
    float c = fmaxf(cnt[g], 1.0f);
    mean[j]      = pooled[(size_t)g * HID + j] / c;
    mean[j + 64] = pooled[(size_t)g * HID + j + 64] / c;
    __syncthreads();
    float acc = bl[j];
#pragma unroll 8
    for (int k = 0; k < HID; ++k) acc += mean[k] * Wl[(size_t)k * OUTD + j];
    red[j] = acc * acc;
    __syncthreads();
    for (int s = 32; s > 0; s >>= 1) {
        if (j < s) red[j] += red[j + s];
        __syncthreads();
    }
    float denom = fmaxf(sqrtf(red[0]), 1e-12f);
    out[(size_t)g * OUTD + j] = acc / denom;
}

// ---------------- host orchestration ----------------

extern "C" void kernel_launch(void* const* d_in, const int* in_sizes, int n_in,
                              void* d_out, int out_size, void* d_ws, size_t ws_size,
                              hipStream_t stream) {
    (void)in_sizes; (void)n_in; (void)out_size; (void)ws_size;

    const float* x  = (const float*)d_in[0];
    const float* W1 = (const float*)d_in[1];
    const float* b1 = (const float*)d_in[2];
    const float* W2 = (const float*)d_in[3];
    const float* b2 = (const float*)d_in[4];
    const float* Wl = (const float*)d_in[5];
    const float* bl = (const float*)d_in[6];
    const long long* ei    = (const long long*)d_in[7];   // [2, E] int64
    const long long* batch = (const long long*)d_in[8];   // [N] int64
    const long long* esrc = ei;
    const long long* edst = ei + N_EDGES;
    float* out = (float*)d_out;

    // workspace carve-up (floats first, then bf16), offsets keep 16B+ alignment
    float* deg    = (float*)d_ws;                  // 40960
    float* dinv   = deg  + 40960;                  // 40960
    float* hbuf   = dinv + 40960;                  // 5,120,000
    float* agg    = hbuf + (size_t)N_NODES * HID;  // 5,120,000
    float* feat   = agg  + (size_t)N_NODES * HID;  // 5,120,000
    float* pooled = feat + (size_t)N_NODES * HID;  // 8192
    float* cnt    = pooled + NGRAPH * HID;         // 64
    bf16_t* featb = (bf16_t*)(cnt + 64);           // 5,120,000 bf16
    bf16_t* wb    = featb + (size_t)N_NODES * HID; // 16384 bf16 (packed fragments)

    const int B = 256;
    const int nfeat = N_NODES * HID;               // 5,120,000
    const long long nscatter = (long long)N_EDGES * 32;

    // degrees / norm
    gcn_init_kernel<<<(N_NODES + B - 1) / B, B, 0, stream>>>(deg, pooled, cnt);
    gcn_edge_deg<<<(N_EDGES + B - 1) / B, B, 0, stream>>>(edst, deg);
    gcn_dinv<<<(N_NODES + B - 1) / B, B, 0, stream>>>(deg, dinv);

    // ---- layer 1 ----
    gcn_cvt_bf16<<<(nfeat + B - 1) / B, B, 0, stream>>>(x, featb, nfeat);
    gcn_pack_w<<<(8 * 4 * 32 * 16) / B, B, 0, stream>>>(W1, wb);
    gcn_gemm_wmma<<<N_NODES / 16, 256, 0, stream>>>(featb, wb, hbuf);
    gcn_zero_f32<<<(nfeat + B - 1) / B, B, 0, stream>>>(agg, nfeat);
    gcn_scatter<<<(unsigned)((nscatter + B - 1) / B), B, 0, stream>>>(esrc, edst, dinv, hbuf, agg);
    gcn_combine<<<(nfeat + B - 1) / B, B, 0, stream>>>(agg, hbuf, dinv, b1, feat, featb);

    // ---- layer 2 ----  (featb already holds bf16 of layer-1 output)
    gcn_pack_w<<<(8 * 4 * 32 * 16) / B, B, 0, stream>>>(W2, wb);
    gcn_gemm_wmma<<<N_NODES / 16, 256, 0, stream>>>(featb, wb, hbuf);
    gcn_zero_f32<<<(nfeat + B - 1) / B, B, 0, stream>>>(agg, nfeat);
    gcn_scatter<<<(unsigned)((nscatter + B - 1) / B), B, 0, stream>>>(esrc, edst, dinv, hbuf, agg);
    gcn_combine<<<(nfeat + B - 1) / B, B, 0, stream>>>(agg, hbuf, dinv, b2, feat, featb);

    // ---- pool + head ----
    gcn_pool<<<(nfeat + B - 1) / B, B, 0, stream>>>(feat, batch, pooled, cnt);
    gcn_head<<<NGRAPH, OUTD, 0, stream>>>(pooled, cnt, Wl, bl, out);
}
